// MultiheadAttention_77103252898183
// MI455X (gfx1250) — compile-verified, matrix operand order checked by reading, not run on hardware
//
#include <hip/hip_runtime.h>
#include <hip/hip_bf16.h>

// ---------------------------------------------------------------------------
// MultiheadAttention forward for MI455X (gfx1250, wave32, WMMA).
// B=4, T=2048, C=1024, H=16, D=64.  All matmuls via v_wmma_f32_16x16x32_f16.
// Staging uses GLOBAL_LOAD_ASYNC_TO_LDS_B128 (ASYNCcnt) where no dtype
// conversion is needed; fp32->fp16 paths use a double-buffered reg->LDS
// pipeline with one barrier per k-step.
// ---------------------------------------------------------------------------

#define DEV __device__ __forceinline__

typedef _Float16 half_t;
typedef __attribute__((ext_vector_type(4)))  _Float16 v4h;
typedef __attribute__((ext_vector_type(8)))  _Float16 v8h;
typedef __attribute__((ext_vector_type(16))) _Float16 v16h;
typedef __attribute__((ext_vector_type(4)))  float    v4f;
typedef __attribute__((ext_vector_type(8)))  float    v8f;

constexpr int   kB    = 4;
constexpr int   kT    = 2048;
constexpr int   kC    = 1024;
constexpr int   kH    = 16;
constexpr int   kD    = 64;
constexpr int   kNTok = kB * kT;          // 8192 rows
constexpr float kScale = 0.125f;          // D^-0.5

union FragU { v16h v; v8h h[2]; };

// Async 16-byte global->LDS copy, tracked by ASYNCcnt (no VGPR data path).
DEV void async_copy_b128(const half_t* lds_dst, const half_t* gsrc) {
  unsigned int loff = (unsigned int)(size_t)lds_dst;   // addr[31:0] = LDS offset
  asm volatile("global_load_async_to_lds_b128 %0, %1, off"
               :
               : "v"(loff), "v"((unsigned long long)(size_t)gsrc)
               : "memory");
}
DEV void wait_asynccnt0() {
  asm volatile("s_wait_asynccnt 0x0" ::: "memory");
}

// Load a 16x32 f16 A-fragment (or mirrored 32x16 B-fragment) per ISA layout:
// lane = g*16 + r; lanes 0-15 hold K 0-7 (VGPR0-3) and 16-23 (VGPR4-7),
// lanes 16-31 hold K 8-15 and 24-31.  Two 16B loads per lane.
DEV v16h load_frag(const half_t* base, int ld) {
  const int lane = threadIdx.x & 31;
  const int r = lane & 15, g = lane >> 4;
  const half_t* p = base + (size_t)r * ld + g * 8;
  FragU u;
  u.h[0] = *(const v8h*)p;         // K = g*8 .. g*8+7
  u.h[1] = *(const v8h*)(p + 16);  // K = 16+g*8 .. 16+g*8+7
  return u.v;
}

// Same fragment, but source is fp32 (softmax weights) -> convert on load.
DEV v16h load_frag_f32(const float* base, size_t ld) {
  const int lane = threadIdx.x & 31;
  const int r = lane & 15, g = lane >> 4;
  const float* p = base + (size_t)r * ld + g * 8;
  v4f x0 = *(const v4f*)p,        x1 = *(const v4f*)(p + 4);
  v4f x2 = *(const v4f*)(p + 16), x3 = *(const v4f*)(p + 20);
  FragU u;
#pragma unroll
  for (int i = 0; i < 4; ++i) {
    u.h[0][i]     = (half_t)x0[i];
    u.h[0][4 + i] = (half_t)x1[i];
    u.h[1][i]     = (half_t)x2[i];
    u.h[1][4 + i] = (half_t)x3[i];
  }
  return u.v;
}

DEV v8f wmma_f16(v16h a, v16h b, v8f c) {
  return __builtin_amdgcn_wmma_f32_16x16x32_f16(
      /*neg_a=*/false, a, /*neg_b=*/false, b,
      /*c_mod=*/(short)0, c, /*reuse_a=*/false, /*reuse_b=*/false);
}

// ---------------------------------------------------------------------------
// Kernel 0: fp32 -> fp16 weight conversion (weights become L2-resident, 8MB).
// ---------------------------------------------------------------------------
__global__ __launch_bounds__(256) void wcvt_kernel(const float* __restrict__ src,
                                                   half_t* __restrict__ dst, int n) {
  int i = (blockIdx.x * 256 + threadIdx.x) * 4;
  if (i < n) {
    v4f f = *(const v4f*)(src + i);
    v4h h;
#pragma unroll
    for (int j = 0; j < 4; ++j) h[j] = (half_t)f[j];
    *(v4h*)(dst + i) = h;
  }
}

// ---------------------------------------------------------------------------
// Kernel 1: Q/K/V projection.  out = (X @ W^T + b) * scale, 8192x1024x1024.
// Block tile 128(M) x 128(N), 8 waves, 16x128 strip per wave.
// Double-buffered LDS pipeline, one barrier per k-step: global loads for
// tile k+1 are issued before the barrier and stored after the WMMAs.
// vmode 0: scatter to [B,H,T,D] (Q,K);  vmode 1: scatter to [B,H,D,T] (V^T).
// ---------------------------------------------------------------------------
__global__ __launch_bounds__(256) void proj_qkv_kernel(
    const float* __restrict__ X, const half_t* __restrict__ W,
    const float* __restrict__ bias, half_t* __restrict__ out,
    int vmode, float scale) {
  __shared__ __align__(16) half_t lds_a[2][128 * 40];
  __shared__ __align__(16) half_t lds_b[2][128 * 40];
  const int tid  = threadIdx.x;
  const int w    = tid >> 5;
  const int row0 = blockIdx.x * 128;
  const int col0 = blockIdx.y * 128;

  v8f acc[8];
#pragma unroll
  for (int cb = 0; cb < 8; ++cb) acc[cb] = (v8f){};

  const int srow = tid >> 1;            // staging row 0..127
  const int skk  = (tid & 1) * 16;      // staging k-offset 0/16
  const float*  abase = X + (size_t)(row0 + srow) * kC + skk;
  const half_t* bbase = W + (size_t)(col0 + srow) * kC + skk;

  v4f f0, f1, f2, f3;                   // in-flight A tile (fp32)
  v8h wb0, wb1;                         // in-flight B tile (fp16)

  auto load_regs = [&](int k0) {
    const float* s = abase + k0;
    f0 = *(const v4f*)s;       f1 = *(const v4f*)(s + 4);
    f2 = *(const v4f*)(s + 8); f3 = *(const v4f*)(s + 12);
    const half_t* sw = bbase + k0;
    wb0 = *(const v8h*)sw;     wb1 = *(const v8h*)(sw + 8);
  };
  auto store_lds = [&](int buf) {
    v8h h0, h1;
#pragma unroll
    for (int i = 0; i < 4; ++i) {
      h0[i] = (half_t)f0[i]; h0[4 + i] = (half_t)f1[i];
      h1[i] = (half_t)f2[i]; h1[4 + i] = (half_t)f3[i];
    }
    *(v8h*)&lds_a[buf][srow * 40 + skk]     = h0;
    *(v8h*)&lds_a[buf][srow * 40 + skk + 8] = h1;
    *(v8h*)&lds_b[buf][srow * 40 + skk]     = wb0;
    *(v8h*)&lds_b[buf][srow * 40 + skk + 8] = wb1;
  };

  load_regs(0);
  store_lds(0);
  for (int it = 0; it < kC / 32; ++it) {
    const int cur = it & 1;
    if (it + 1 < kC / 32) load_regs((it + 1) * 32);  // overlap with WMMA
    __syncthreads();
    v16h a = load_frag(&lds_a[cur][w * 16 * 40], 40);
#pragma unroll
    for (int cb = 0; cb < 8; ++cb) {
      v16h bf = load_frag(&lds_b[cur][cb * 16 * 40], 40);
      acc[cb] = wmma_f16(a, bf, acc[cb]);
    }
    if (it + 1 < kC / 32) store_lds(1 - cur);
  }

  // Epilogue: bias, scale, scatter into head layout.
  const int lane = tid & 31, g = lane >> 4, nn = lane & 15;
#pragma unroll
  for (int cb = 0; cb < 8; ++cb) {
#pragma unroll
    for (int r = 0; r < 8; ++r) {
      const int grow = row0 + w * 16 + r + g * 8;   // token index
      const int gcol = col0 + cb * 16 + nn;         // feature index
      float val = (acc[cb][r] + bias[gcol]) * scale;
      const int bb = grow >> 11, t = grow & (kT - 1);
      const int hh = gcol >> 6,  d = gcol & (kD - 1);
      size_t dst;
      if (vmode == 0) dst = (((size_t)(bb * kH + hh) * kT + t) * kD) + d;
      else            dst = (((size_t)(bb * kH + hh) * kD + d) * kT) + t;
      out[dst] = (half_t)val;
    }
  }
}

// ---------------------------------------------------------------------------
// Kernel 2: scores S = Q K^T with causal + key-padding mask (fp32 out).
// One block per 128x128 tile of one head.  K tile staged in LDS via
// async global->LDS (4x b128 per thread), fully-masked tiles early-out.
// ---------------------------------------------------------------------------
__global__ __launch_bounds__(256) void scores_kernel(
    const half_t* __restrict__ qh, const half_t* __restrict__ kh,
    const unsigned char* __restrict__ kp, float* __restrict__ attn) {
  __shared__ __align__(16) half_t lds_k[128 * 72];
  const int tid = threadIdx.x;
  const int w   = tid >> 5;
  const int i0  = blockIdx.x * 128;
  const int j0  = blockIdx.y * 128;
  const int bh  = blockIdx.z;
  const int bb  = bh >> 4;
  const float kNegInf = -__builtin_inff();

  if (j0 >= i0 + 128) {  // tile entirely above the diagonal: all masked
    for (int e = tid; e < 128 * 128; e += 256) {
      const int r = e >> 7, c = e & 127;
      attn[((size_t)bh * kT + (i0 + r)) * kT + (j0 + c)] = kNegInf;
    }
    return;
  }

  const half_t* qb = qh + (size_t)bh * kT * kD;
  const half_t* kb = kh + (size_t)bh * kT * kD;

  // Async-stage K tile: 128 rows x 64 halfs; 32 halfs (4x b128) per thread.
  {
    const int srow = tid >> 1;
    const int skk  = (tid & 1) * 32;
    const half_t* src = kb + (size_t)(j0 + srow) * kD + skk;
#pragma unroll
    for (int q = 0; q < 4; ++q)
      async_copy_b128(&lds_k[srow * 72 + skk + q * 8], src + q * 8);
  }
  wait_asynccnt0();
  __syncthreads();

  v8f acc[8];
#pragma unroll
  for (int cb = 0; cb < 8; ++cb) acc[cb] = (v8f){};

#pragma unroll
  for (int k0 = 0; k0 < kD; k0 += 32) {
    v16h a = load_frag(qb + (size_t)(i0 + w * 16) * kD + k0, kD);
#pragma unroll
    for (int cb = 0; cb < 8; ++cb) {
      v16h bf = load_frag(lds_k + cb * 16 * 72 + k0, 72);
      acc[cb] = wmma_f16(a, bf, acc[cb]);
    }
  }

  const int lane = tid & 31, g = lane >> 4, nn = lane & 15;
#pragma unroll
  for (int cb = 0; cb < 8; ++cb) {
    const int gj = j0 + cb * 16 + nn;
    const bool pad = kp[bb * kT + gj] != 0;
#pragma unroll
    for (int r = 0; r < 8; ++r) {
      const int gi = i0 + w * 16 + r + g * 8;
      const bool masked = (gj > gi) || pad;
      attn[((size_t)bh * kT + gi) * kT + gj] = masked ? kNegInf : acc[cb][r];
    }
  }
}

// ---------------------------------------------------------------------------
// Kernel 3: in-place row softmax over T=2048.  One block per row.
// ---------------------------------------------------------------------------
__global__ __launch_bounds__(256) void softmax_kernel(float* __restrict__ attn) {
  __shared__ float red[256];
  const size_t row = blockIdx.x;
  float* p = attn + row * (size_t)kT;
  const int t = threadIdx.x;
  const float kNegInf = -__builtin_inff();

  float m = kNegInf;
  for (int j = t; j < kT; j += 256) m = fmaxf(m, p[j]);
  red[t] = m;
  __syncthreads();
  for (int s = 128; s > 0; s >>= 1) {
    if (t < s) red[t] = fmaxf(red[t], red[t + s]);
    __syncthreads();
  }
  m = red[0];
  __syncthreads();

  float sum = 0.f;
  for (int j = t; j < kT; j += 256) {
    float e = (m == kNegInf) ? 0.f : __expf(p[j] - m);
    p[j] = e;
    sum += e;
  }
  red[t] = sum;
  __syncthreads();
  for (int s = 128; s > 0; s >>= 1) {
    if (t < s) red[t] += red[t + s];
    __syncthreads();
  }
  const float inv = (red[0] > 0.f) ? (1.f / red[0]) : 0.f;
  for (int j = t; j < kT; j += 256) p[j] *= inv;
}

// ---------------------------------------------------------------------------
// Kernel 4: O = P V.  P fp32 -> fp16 fragments on the fly (dominant HBM
// stream, ~1GB).  V chunk (64 x 32 halfs) staged in LDS per k-step via
// async global->LDS, double-buffered so the copy of chunk k+1 overlaps the
// WMMAs of chunk k; shared by all 8 waves (8x traffic reduction).
// Block: 128 rows x 64 cols of one head.
// ---------------------------------------------------------------------------
__global__ __launch_bounds__(256) void av_kernel(
    const float* __restrict__ attn, const half_t* __restrict__ vth,
    half_t* __restrict__ oh) {
  __shared__ __align__(16) half_t lds_v[2][64 * 40];
  const int tid = threadIdx.x;
  const int w   = tid >> 5;
  const int i0  = blockIdx.x * 128;
  const int bh  = blockIdx.z;

  const float*  prow = attn + ((size_t)bh * kT + (i0 + w * 16)) * kT;
  const half_t* vb   = vth + (size_t)bh * kD * kT;

  const int vrow = tid >> 2;            // 0..63 (d)
  const int vkk  = (tid & 3) * 8;       // 0/8/16/24 halfs within 32-chunk
  auto stage_v = [&](int k0, int buf) { // one async b128 per thread
    async_copy_b128(&lds_v[buf][vrow * 40 + vkk],
                    vb + (size_t)vrow * kT + k0 + vkk);
  };

  v8f acc[4];
#pragma unroll
  for (int cb = 0; cb < 4; ++cb) acc[cb] = (v8f){};

  stage_v(0, 0);
  for (int it = 0; it < kT / 32; ++it) {
    const int cur = it & 1;
    wait_asynccnt0();                    // our stage(it) landed
    __syncthreads();                     // everyone's stage(it) landed,
                                         // everyone done reading buf[1-cur]
    if (it + 1 < kT / 32) stage_v((it + 1) * 32, 1 - cur);
    v16h a = load_frag_f32(prow + it * 32, kT);
#pragma unroll
    for (int cb = 0; cb < 4; ++cb) {
      v16h bf = load_frag(&lds_v[cur][cb * 16 * 40], 40);
      acc[cb] = wmma_f16(a, bf, acc[cb]);
    }
  }

  const int lane = tid & 31, g = lane >> 4, nn = lane & 15;
  const int bb = bh >> 4, hh = bh & 15;
#pragma unroll
  for (int cb = 0; cb < 4; ++cb) {
#pragma unroll
    for (int r = 0; r < 8; ++r) {
      const int gi = i0 + w * 16 + r + g * 8;
      const int gd = cb * 16 + nn;
      oh[(size_t)(bb * kT + gi) * kC + hh * kD + gd] = (half_t)acc[cb][r];
    }
  }
}

// ---------------------------------------------------------------------------
// Kernel 5: output projection y = Oh @ Wo^T + bo (fp16 A, fp32 out).
// Same double-buffered single-barrier pipeline as proj_qkv.
// ---------------------------------------------------------------------------
__global__ __launch_bounds__(256) void outproj_kernel(
    const half_t* __restrict__ A, const half_t* __restrict__ W,
    const float* __restrict__ bias, float* __restrict__ y) {
  __shared__ __align__(16) half_t lds_a[2][128 * 40];
  __shared__ __align__(16) half_t lds_b[2][128 * 40];
  const int tid  = threadIdx.x;
  const int w    = tid >> 5;
  const int row0 = blockIdx.x * 128;
  const int col0 = blockIdx.y * 128;

  v8f acc[8];
#pragma unroll
  for (int cb = 0; cb < 8; ++cb) acc[cb] = (v8f){};

  const int srow = tid >> 1;
  const int skk  = (tid & 1) * 16;
  const half_t* abase = A + (size_t)(row0 + srow) * kC + skk;
  const half_t* bbase = W + (size_t)(col0 + srow) * kC + skk;

  v8h a0, a1, wb0, wb1;
  auto load_regs = [&](int k0) {
    a0  = *(const v8h*)(abase + k0); a1  = *(const v8h*)(abase + k0 + 8);
    wb0 = *(const v8h*)(bbase + k0); wb1 = *(const v8h*)(bbase + k0 + 8);
  };
  auto store_lds = [&](int buf) {
    *(v8h*)&lds_a[buf][srow * 40 + skk]     = a0;
    *(v8h*)&lds_a[buf][srow * 40 + skk + 8] = a1;
    *(v8h*)&lds_b[buf][srow * 40 + skk]     = wb0;
    *(v8h*)&lds_b[buf][srow * 40 + skk + 8] = wb1;
  };

  load_regs(0);
  store_lds(0);
  for (int it = 0; it < kC / 32; ++it) {
    const int cur = it & 1;
    if (it + 1 < kC / 32) load_regs((it + 1) * 32);
    __syncthreads();
    v16h a = load_frag(&lds_a[cur][w * 16 * 40], 40);
#pragma unroll
    for (int cb = 0; cb < 8; ++cb) {
      v16h bf = load_frag(&lds_b[cur][cb * 16 * 40], 40);
      acc[cb] = wmma_f16(a, bf, acc[cb]);
    }
    if (it + 1 < kC / 32) store_lds(1 - cur);
  }

  const int lane = tid & 31, g = lane >> 4, nn = lane & 15;
#pragma unroll
  for (int cb = 0; cb < 8; ++cb) {
#pragma unroll
    for (int r = 0; r < 8; ++r) {
      const int grow = row0 + w * 16 + r + g * 8;
      const int gcol = col0 + cb * 16 + nn;
      y[(size_t)grow * kC + gcol] = acc[cb][r] + bias[gcol];
    }
  }
}

// ---------------------------------------------------------------------------
// Host-side orchestration.
// ---------------------------------------------------------------------------
extern "C" void kernel_launch(void* const* d_in, const int* in_sizes, int n_in,
                              void* d_out, int out_size, void* d_ws, size_t ws_size,
                              hipStream_t stream) {
  (void)in_sizes; (void)n_in; (void)out_size; (void)ws_size;

  const float* query = (const float*)d_in[0];
  const float* key   = (const float*)d_in[1];
  const float* value = (const float*)d_in[2];
  // d_in[3] = attn_mask: causal triu(k=1); computed analytically (j > i).
  const unsigned char* kp = (const unsigned char*)d_in[4];
  const float* Wq = (const float*)d_in[5];  const float* bq = (const float*)d_in[6];
  const float* Wk = (const float*)d_in[7];  const float* bk = (const float*)d_in[8];
  const float* Wv = (const float*)d_in[9];  const float* bv = (const float*)d_in[10];
  const float* Wo = (const float*)d_in[11]; const float* bo = (const float*)d_in[12];

  float* y    = (float*)d_out;                 // [B,T,C]
  float* attn = y + (size_t)kB * kT * kC;      // [B,H,T,T] fp32

  // Workspace layout (fp16 halves): ~75.5 MB total.
  half_t* ws  = (half_t*)d_ws;
  half_t* qh  = ws;                                  // [B,H,T,D]
  half_t* khb = qh  + (size_t)kNTok * kC;            // [B,H,T,D]
  half_t* vth = khb + (size_t)kNTok * kC;            // [B,H,D,T]
  half_t* oh  = vth + (size_t)kNTok * kC;            // [B,T,C]
  half_t* wqh = oh  + (size_t)kNTok * kC;            // [C,C] each
  half_t* wkh = wqh + (size_t)kC * kC;
  half_t* wvh = wkh + (size_t)kC * kC;
  half_t* woh = wvh + (size_t)kC * kC;

  const int nW = kC * kC;
  wcvt_kernel<<<nW / 1024, 256, 0, stream>>>(Wq, wqh, nW);
  wcvt_kernel<<<nW / 1024, 256, 0, stream>>>(Wk, wkh, nW);
  wcvt_kernel<<<nW / 1024, 256, 0, stream>>>(Wv, wvh, nW);
  wcvt_kernel<<<nW / 1024, 256, 0, stream>>>(Wo, woh, nW);

  dim3 gProj(kNTok / 128, kC / 128);  // (64, 8)
  proj_qkv_kernel<<<gProj, 256, 0, stream>>>(query, wqh, bq, qh, 0, kScale);
  proj_qkv_kernel<<<gProj, 256, 0, stream>>>(key,   wkh, bk, khb, 0, 1.0f);
  proj_qkv_kernel<<<gProj, 256, 0, stream>>>(value, wvh, bv, vth, 1, 1.0f);

  dim3 gScore(kT / 128, kT / 128, kB * kH);  // (16, 16, 64)
  scores_kernel<<<gScore, 256, 0, stream>>>(qh, khb, kp, attn);

  softmax_kernel<<<(unsigned)((size_t)kB * kH * kT), 256, 0, stream>>>(attn);

  dim3 gAV(kT / 128, 1, kB * kH);  // (16, 1, 64)
  av_kernel<<<gAV, 256, 0, stream>>>(attn, vth, oh);

  outproj_kernel<<<gProj, 256, 0, stream>>>(oh, woh, bo, y);
}
